// Beit3MultiheadAttention_46523085750415
// MI455X (gfx1250) — compile-verified, matrix operand order checked by reading, not run on hardware
//
#include <hip/hip_runtime.h>
#include <hip/hip_bf16.h>
#include <math.h>

typedef __attribute__((ext_vector_type(16))) _Float16 v16h;
typedef __attribute__((ext_vector_type(8)))  _Float16 h8;
typedef __attribute__((ext_vector_type(4)))  _Float16 h4;
typedef __attribute__((ext_vector_type(8)))  float    v8f;

#define EMBED   768
#define HEADS   12
#define HD      64
#define BB      2
#define TT      2048
#define SS      2048
#define SCALING 0.125f   /* 64^-0.5 */
#define LN_EPS  1e-5f

// ---------- wave32 reductions ----------
static __device__ __forceinline__ float wave_max(float v) {
    #pragma unroll
    for (int o = 16; o > 0; o >>= 1) v = fmaxf(v, __shfl_xor(v, o, 32));
    return v;
}
static __device__ __forceinline__ float wave_sum(float v) {
    #pragma unroll
    for (int o = 16; o > 0; o >>= 1) v += __shfl_xor(v, o, 32);
    return v;
}

// Build a 16xf16 WMMA operand from two contiguous 8-half runs at p and p+16.
// (ISA 16-bit A/B layout: lane holds K = koff..koff+7 and koff+16..koff+23.)
static __device__ __forceinline__ v16h make_ab_f16(const _Float16* p) {
    h8 lo = *(const h8*)p;
    h8 hi = *(const h8*)(p + 16);
    v16h r;
    #pragma unroll
    for (int i = 0; i < 8; ++i) { r[i] = lo[i]; r[8 + i] = hi[i]; }
    return r;
}

// =====================================================================
// Kernel 0: one-shot f32 -> f16 conversion of the four weight matrices.
// grid=(576,4), block=256; each thread converts one float4.
// =====================================================================
__global__ __launch_bounds__(256) void cvt_w_kernel(
    const float* __restrict__ Wq, const float* __restrict__ Wk,
    const float* __restrict__ Wv, const float* __restrict__ Wo,
    _Float16* __restrict__ w16)   // 4 matrices back to back
{
    const int which = blockIdx.y;
    const float* W = (which == 0) ? Wq : (which == 1) ? Wk : (which == 2) ? Wv : Wo;
    _Float16* dst = w16 + (size_t)which * EMBED * EMBED;
    const int i4 = blockIdx.x * 256 + threadIdx.x;          // float4 index
    const float4 v = ((const float4*)W)[i4];
    h4 o; o[0] = (_Float16)v.x; o[1] = (_Float16)v.y; o[2] = (_Float16)v.z; o[3] = (_Float16)v.w;
    *(h4*)&dst[(size_t)i4 * 4] = o;
}

// =====================================================================
// Kernel 1: fused Q/K/V projection.  grid=(64,3), block=256 (8 waves).
// blockIdx.y: 0->Q (scaled), 1->K, 2->V (stored transposed [b,h,hd,s]).
// LDS: staged input tile 64x776 f16 (padded row to avoid bank conflicts).
// B operand comes from pre-converted f16 weights (pure 16B loads).
// =====================================================================
__global__ __launch_bounds__(256) void qkv_gemm_kernel(
    const float* __restrict__ Xq, const float* __restrict__ Xk, const float* __restrict__ Xv,
    const _Float16* __restrict__ w16,
    const float* __restrict__ bq, const float* __restrict__ bk, const float* __restrict__ bv,
    _Float16* __restrict__ q16, _Float16* __restrict__ k16, _Float16* __restrict__ vT16)
{
    extern __shared__ _Float16 lds[];            // 64 x 776 halves
    const int SP    = 776;
    const int M0    = blockIdx.x * 64;
    const int which = blockIdx.y;
    const int tid   = threadIdx.x;

    const float* X        = (which == 0) ? Xq : (which == 1) ? Xk : Xv;
    const _Float16* W     = w16 + (size_t)which * EMBED * EMBED;
    const float* bias     = (which == 0) ? bq : (which == 1) ? bk : bv;
    const float scale     = (which == 0) ? SCALING : 1.0f;

    for (int idx = tid; idx < 64 * 768; idx += 256) {
        int r = idx / 768, c = idx - r * 768;
        lds[r * SP + c] = (_Float16)X[(size_t)(M0 + r) * 768 + c];
    }
    __syncthreads();

    const int lane = tid & 31, wid = tid >> 5;
    const int mrow = lane & 15;              // A row / B col inside tile
    const int koff = (lane >> 4) * 8;        // K sub-offset per lane half

    for (int nt = wid * 6; nt < wid * 6 + 6; ++nt) {
        const int n0 = nt * 16;
        const _Float16* wrow = W + (size_t)(n0 + mrow) * 768;  // B col n = W row n (out = X @ W^T)
        v8f acc[4] = {};
        for (int ks = 0; ks < 24; ++ks) {
            const int k0 = ks * 32;
            v16h bm = make_ab_f16(wrow + k0 + koff);
            #pragma unroll
            for (int mt = 0; mt < 4; ++mt) {
                v16h am = make_ab_f16(&lds[(mt * 16 + mrow) * SP + k0 + koff]);
                acc[mt] = __builtin_amdgcn_wmma_f32_16x16x32_f16(
                    false, am, false, bm, (short)0, acc[mt], false, false);
            }
        }
        const int gn = n0 + mrow;            // global output column
        const int h  = gn >> 6, hd = gn & 63;
        const float bval = bias[gn];
        #pragma unroll
        for (int mt = 0; mt < 4; ++mt) {
            const int mbase = M0 + mt * 16 + (lane >> 4) * 8;  // C: rows j / j+8
            if (which == 2) {
                // V transposed: 8 consecutive s per lane -> one 16B store
                int b = mbase / SS, s = mbase - b * SS;
                h8 vv;
                #pragma unroll
                for (int j = 0; j < 8; ++j) vv[j] = (_Float16)(acc[mt][j] * scale + bval);
                *(h8*)&vT16[((size_t)(b * HEADS + h) * HD + hd) * SS + s] = vv;
            } else {
                _Float16* dst = (which == 0) ? q16 : k16;
                #pragma unroll
                for (int j = 0; j < 8; ++j) {
                    int gm = mbase + j;
                    int b = gm / TT, t = gm - b * TT;
                    dst[((size_t)(b * HEADS + h) * TT + t) * HD + hd] =
                        (_Float16)(acc[mt][j] * scale + bval);
                }
            }
        }
    }
}

// =====================================================================
// Kernel 2: scores + softmax + weights write + P@V.
// grid=(T/16, H, B), block=256 (8 waves).
// LDS: 16 x 2052 f32 score slab (131 KB, fits 320KB WGP LDS) + 4KB partials.
// =====================================================================
__global__ __launch_bounds__(256) void attn_kernel(
    const _Float16* __restrict__ q16, const _Float16* __restrict__ k16,
    const _Float16* __restrict__ vT16,
    const float* __restrict__ attn_mask, const int* __restrict__ kpm,
    const float* __restrict__ relpos,
    float* __restrict__ weights_out, float* __restrict__ attn_pre)
{
    extern __shared__ float sc[];                 // [16][2052] + partials
    const int SP2 = 2052;
    float* part = sc + 16 * SP2;

    const int t0 = blockIdx.x * 16;
    const int h  = blockIdx.y, b = blockIdx.z;
    const size_t bh = (size_t)(b * HEADS + h);

    const int tid = threadIdx.x, lane = tid & 31, wid = tid >> 5;
    const int mrow = lane & 15;
    const int koff = (lane >> 4) * 8;
    const int mhalf = (lane >> 4) * 8;

    // Q tile A-operands (held in registers for the whole score sweep)
    const _Float16* qrow = q16 + (bh * TT + t0 + mrow) * HD;
    const v16h A0 = make_ab_f16(qrow + 0  + koff);
    const v16h A1 = make_ab_f16(qrow + 32 + koff);

    // ---- score phase: each wave covers s in [wid*256, wid*256+256) ----
    for (int st = 0; st < 16; ++st) {
        const int s0 = wid * 256 + st * 16;
        const _Float16* krow = k16 + (bh * SS + s0 + mrow) * HD;
        v8f acc = {};
        v16h B0 = make_ab_f16(krow + 0  + koff);
        v16h B1 = make_ab_f16(krow + 32 + koff);
        acc = __builtin_amdgcn_wmma_f32_16x16x32_f16(false, A0, false, B0, (short)0, acc, false, false);
        acc = __builtin_amdgcn_wmma_f32_16x16x32_f16(false, A1, false, B1, (short)0, acc, false, false);

        const int s = s0 + mrow;
        const int masked = kpm[b * SS + s];
        // prefetch next tile's rel_pos row chunk (global_prefetch_b8)
        __builtin_prefetch(&relpos[(bh * TT + t0 + mhalf) * SS + s0 + 16], 0, 0);
        #pragma unroll
        for (int j = 0; j < 8; ++j) {
            const int tl = j + mhalf;
            const int gt = t0 + tl;
            float v = acc[j];
            if (__builtin_isnan(v)) v = 0.0f;                       // nan_to_num
            v += attn_mask[(size_t)gt * SS + s];
            if (masked) v = -__builtin_inff();                      // key padding
            v += __builtin_nontemporal_load(&relpos[(bh * TT + gt) * SS + s]);
            sc[tl * SP2 + s] = v;
        }
    }
    __syncthreads();

    // ---- softmax: wave w handles rows w and w+8 ----
    for (int r = wid; r < 16; r += 8) {
        float* row = sc + r * SP2;
        float m = -__builtin_inff();
        for (int s = lane; s < SS; s += 32) m = fmaxf(m, row[s]);
        m = wave_max(m);
        float sum = 0.0f;
        for (int s = lane; s < SS; s += 32) {
            float e = __expf(row[s] - m);
            row[s] = e;
            sum += e;
        }
        sum = wave_sum(sum);
        const float inv = 1.0f / sum;
        float* wout = weights_out + (((size_t)h * BB + b) * TT + t0 + r) * SS;
        for (int s = lane; s < SS; s += 32) {
            float p = row[s] * inv;
            row[s] = p;                                   // keep for P@V
            __builtin_nontemporal_store(p, &wout[s]);     // stream weights out
        }
    }
    __syncthreads();

    // ---- P@V: 4 hd-tiles x 2 K-halves across 8 waves ----
    const int n0    = (wid & 3) * 16;
    const int khalf = wid >> 2;
    const _Float16* vrow = vT16 + (bh * HD + n0 + mrow) * SS;  // B col n contiguous in s
    v8f acc = {};
    for (int ks = 0; ks < 32; ++ks) {
        const int k0 = khalf * 1024 + ks * 32;
        const float* arow = sc + mrow * SP2 + k0 + koff;
        v16h a;
        #pragma unroll
        for (int i = 0; i < 8; ++i) { a[i] = (_Float16)arow[i]; a[8 + i] = (_Float16)arow[16 + i]; }
        v16h bm = make_ab_f16(vrow + k0 + koff);
        acc = __builtin_amdgcn_wmma_f32_16x16x32_f16(false, a, false, bm, (short)0, acc, false, false);
    }
    if (khalf == 1) {
        #pragma unroll
        for (int j = 0; j < 8; ++j) part[(wid - 4) * 256 + j * 32 + lane] = acc[j];
    }
    __syncthreads();
    if (khalf == 0) {
        #pragma unroll
        for (int j = 0; j < 8; ++j) {
            float v = acc[j] + part[wid * 256 + j * 32 + lane];
            const int tl = j + mhalf;
            attn_pre[((size_t)b * TT + t0 + tl) * EMBED + h * HD + n0 + mrow] = v;
        }
    }
}

// =====================================================================
// Kernel 3: LayerNorm (wave per row) -> f16
// =====================================================================
__global__ __launch_bounds__(256) void ln_kernel(
    const float* __restrict__ x, const float* __restrict__ g,
    const float* __restrict__ bta, _Float16* __restrict__ y)
{
    const int row  = blockIdx.x * 8 + (threadIdx.x >> 5);
    const int lane = threadIdx.x & 31;
    const float* xr = x + (size_t)row * EMBED;
    float s = 0.0f, s2 = 0.0f;
    for (int c = lane; c < EMBED; c += 32) { float v = xr[c]; s += v; s2 += v * v; }
    s  = wave_sum(s);
    s2 = wave_sum(s2);
    const float mu  = s * (1.0f / EMBED);
    const float var = s2 * (1.0f / EMBED) - mu * mu;
    const float rs  = rsqrtf(var + LN_EPS);
    _Float16* yr = y + (size_t)row * EMBED;
    for (int c = lane; c < EMBED; c += 32)
        yr[c] = (_Float16)((xr[c] - mu) * rs * g[c] + bta[c]);
}

// =====================================================================
// Kernel 4: output projection GEMM (f16 staged input, f16 weights) -> f32
// =====================================================================
__global__ __launch_bounds__(256) void out_gemm_kernel(
    const _Float16* __restrict__ X16, const _Float16* __restrict__ Wo16,
    const float* __restrict__ bo, float* __restrict__ out)
{
    extern __shared__ _Float16 lds[];            // 64 x 776 halves
    const int SP  = 776;
    const int M0  = blockIdx.x * 64;
    const int tid = threadIdx.x, lane = tid & 31, wid = tid >> 5;

    for (int idx = tid; idx < 64 * 768; idx += 256) {
        int r = idx / 768, c = idx - r * 768;
        lds[r * SP + c] = X16[(size_t)(M0 + r) * 768 + c];
    }
    __syncthreads();

    const int mrow = lane & 15, koff = (lane >> 4) * 8;
    for (int nt = wid * 6; nt < wid * 6 + 6; ++nt) {
        const int n0 = nt * 16;
        const _Float16* wrow = Wo16 + (size_t)(n0 + mrow) * 768;
        v8f acc[4] = {};
        for (int ks = 0; ks < 24; ++ks) {
            const int k0 = ks * 32;
            v16h bm = make_ab_f16(wrow + k0 + koff);
            #pragma unroll
            for (int mt = 0; mt < 4; ++mt) {
                v16h am = make_ab_f16(&lds[(mt * 16 + mrow) * SP + k0 + koff]);
                acc[mt] = __builtin_amdgcn_wmma_f32_16x16x32_f16(
                    false, am, false, bm, (short)0, acc[mt], false, false);
            }
        }
        const float bval = bo[n0 + mrow];
        #pragma unroll
        for (int mt = 0; mt < 4; ++mt) {
            const int mbase = M0 + mt * 16 + (lane >> 4) * 8;
            #pragma unroll
            for (int j = 0; j < 8; ++j)
                out[(size_t)(mbase + j) * 768 + n0 + mrow] = acc[mt][j] + bval;
        }
    }
}

// =====================================================================
extern "C" void kernel_launch(void* const* d_in, const int* in_sizes, int n_in,
                              void* d_out, int out_size, void* d_ws, size_t ws_size,
                              hipStream_t stream)
{
    const float* query = (const float*)d_in[0];
    const float* key   = (const float*)d_in[1];
    const float* value = (const float*)d_in[2];
    const int*   kpm   = (const int*)  d_in[3];
    const float* amask = (const float*)d_in[4];
    const float* relp  = (const float*)d_in[5];
    const float* Wq = (const float*)d_in[6];  const float* bq = (const float*)d_in[7];
    const float* Wk = (const float*)d_in[8];  const float* bk = (const float*)d_in[9];
    const float* Wv = (const float*)d_in[10]; const float* bv = (const float*)d_in[11];
    const float* Wo = (const float*)d_in[12]; const float* bo = (const float*)d_in[13];
    const float* lng = (const float*)d_in[14]; const float* lnb = (const float*)d_in[15];

    char* ws = (char*)d_ws;
    size_t off = 0;
    _Float16* q16  = (_Float16*)(ws + off); off += (size_t)BB * HEADS * TT * HD * 2;
    _Float16* k16  = (_Float16*)(ws + off); off += (size_t)BB * HEADS * SS * HD * 2;
    _Float16* vT16 = (_Float16*)(ws + off); off += (size_t)BB * HEADS * SS * HD * 2;
    float*    attn_pre = (float*)(ws + off); off += (size_t)BB * TT * EMBED * 4;
    _Float16* ln16 = (_Float16*)(ws + off); off += (size_t)BB * TT * EMBED * 2;
    _Float16* w16  = (_Float16*)(ws + off); off += (size_t)4 * EMBED * EMBED * 2;

    float* out_attn = (float*)d_out;                               // [B,T,768]
    float* out_w    = out_attn + (size_t)BB * TT * EMBED;          // [H,B,T,S]

    const size_t lds_gemm = (size_t)64 * 776 * sizeof(_Float16);   // 99,328 B
    const size_t lds_attn = (size_t)(16 * 2052 + 4 * 256) * 4;     // 135,424 B

    cvt_w_kernel<<<dim3((EMBED * EMBED / 4) / 256, 4), 256, 0, stream>>>(
        Wq, Wk, Wv, Wo, w16);

    qkv_gemm_kernel<<<dim3((BB * TT) / 64, 3), 256, lds_gemm, stream>>>(
        query, key, value, w16, bq, bk, bv, q16, k16, vT16);

    attn_kernel<<<dim3(TT / 16, HEADS, BB), 256, lds_attn, stream>>>(
        q16, k16, vT16, amask, kpm, relp, out_w, attn_pre);

    ln_kernel<<<dim3((BB * TT) / 8), 256, 0, stream>>>(attn_pre, lng, lnb, ln16);

    out_gemm_kernel<<<dim3((BB * TT) / 64), 256, lds_gemm, stream>>>(
        ln16, w16 + (size_t)3 * EMBED * EMBED, bo, out_attn);
}
